// MambaSE_78554951844226
// MI455X (gfx1250) — matrix operand order repaired and checked by reading, not compiled, not run on hardware
//
#include <hip/hip_runtime.h>
#include <hip/hip_bf16.h>
#include <cstdint>
#include <cstddef>

#define DM   768            // D_MODEL
#define DI   1536           // D_INNER
#define DS   16             // D_STATE
#define DTR  48             // DT_RANK
#define KPAD 64             // padded dt-GEMM K
#define XDL  96             // x_dbl padded leading dim (80 valid cols)
#define NXP  80             // x_proj rows (valid)
#define NXPP 128            // x_proj rows (padded to N-block)
#define NB   16             // BATCH
#define SL   2048           // SEQ
#define BL   (NB * SL)      // 32768 rows
#define E2   (2 * DI)       // 3072
#define BN_EPS 1e-5f

typedef __attribute__((ext_vector_type(16))) __bf16 v16bf;
typedef __attribute__((ext_vector_type(8)))  __bf16 v8bf;
typedef __attribute__((ext_vector_type(8)))  float  v8f;

__device__ __forceinline__ float bf2f(__bf16 b) {
  unsigned short h = __builtin_bit_cast(unsigned short, b);
  unsigned u = ((unsigned)h) << 16;
  return __builtin_bit_cast(float, u);
}
__device__ __forceinline__ __bf16 f2bf(float f) {
  unsigned u = __builtin_bit_cast(unsigned, f);
  unsigned r = u + 0x7FFFu + ((u >> 16) & 1u);   // round-to-nearest-even
  unsigned short h = (unsigned short)(r >> 16);
  return __builtin_bit_cast(__bf16, h);
}

// ---------------------------------------------------------------- converts
__global__ __launch_bounds__(256) void cvt_f32_bf16(const float* __restrict__ in,
                                                    __bf16* __restrict__ out, size_t n) {
  size_t i = (size_t)blockIdx.x * 256 + threadIdx.x;
  if (i < n) out[i] = f2bf(in[i]);
}

// dt_proj_w (DI x 48) -> (DI x 64), cols 48..63 zeroed so padded-K GEMM is exact
__global__ __launch_bounds__(256) void cvt_pad_dtw(const float* __restrict__ in,
                                                   __bf16* __restrict__ out) {
  int idx = blockIdx.x * 256 + threadIdx.x;       // DI*64 = 98304 threads
  int r = idx >> 6, c = idx & 63;
  out[idx] = (c < DTR) ? f2bf(in[r * DTR + c]) : f2bf(0.0f);
}

// x_proj_w (80 x DI) -> (128 x DI), rows 80..127 zeroed: lets the GEMM drop
// all N-bounds checks from the inner loop (cols >= 80 are never stored).
__global__ __launch_bounds__(256) void cvt_pad_xpw(const float* __restrict__ in,
                                                   __bf16* __restrict__ out) {
  int idx = blockIdx.x * 256 + threadIdx.x;       // NXPP*DI threads
  int r = idx / DI;
  out[idx] = (r < NXP) ? f2bf(in[idx]) : f2bf(0.0f);
}

// ---------------------------------------------------------------- WMMA GEMM
// C(MxN) = A(MxK, row-major bf16) * W(NxK, row-major bf16)^T
// W must be readable (padded) up to the 128-aligned N block bound.
// EPI: 0 = store bf16, 1 = store f32, 2 = softplus(acc + bias[col]) -> bf16
template <int EPI>
__global__ __launch_bounds__(256) void gemm_bf16_wmma(
    const __bf16* __restrict__ A, const __bf16* __restrict__ W,
    void* __restrict__ Cout, const float* __restrict__ bias,
    int M, int N, int K, int lda, int ldb, int ldc) {
  const int lane = threadIdx.x & 31;
  const int wave = threadIdx.x >> 5;   // 8 waves: 4 along M x 2 along N
  const int wm   = wave & 3;
  const int wn   = wave >> 2;
  const int m0   = blockIdx.x * 128 + wm * 32;   // wave rows m0..m0+31
  const int n0   = blockIdx.y * 128 + wn * 64;   // wave cols n0..n0+63
  const int l16  = lane & 15;
  const int half = lane >> 4;

  v8f acc[2][4];
#pragma unroll
  for (int i = 0; i < 2; ++i)
#pragma unroll
    for (int j = 0; j < 4; ++j)
#pragma unroll
      for (int r = 0; r < 8; ++r) acc[i][j][r] = 0.0f;

  const __bf16* a0p = A + (size_t)(m0 + l16) * lda;
  const __bf16* a1p = A + (size_t)(m0 + 16 + l16) * lda;
  const __bf16* b0p = W + (size_t)(n0 + l16) * ldb + half * 16;
  const __bf16* b1p = b0p + (size_t)16 * ldb;
  const __bf16* b2p = b0p + (size_t)32 * ldb;
  const __bf16* b3p = b0p + (size_t)48 * ldb;

  for (int k0 = 0; k0 < K; k0 += 32) {
    // A fragment (16x32 bf16): lane = M row; elems e -> K = half*16 + e
    // (same contiguous K order as B)
    v8bf a0lo = *(const v8bf*)(a0p + k0 + half * 16);
    v8bf a0hi = *(const v8bf*)(a0p + k0 + half * 16 + 8);
    v8bf a1lo = *(const v8bf*)(a1p + k0 + half * 16);
    v8bf a1hi = *(const v8bf*)(a1p + k0 + half * 16 + 8);
    v16bf afr[2];
    afr[0] = __builtin_shufflevector(a0lo, a0hi, 0,1,2,3,4,5,6,7,8,9,10,11,12,13,14,15);
    afr[1] = __builtin_shufflevector(a1lo, a1hi, 0,1,2,3,4,5,6,7,8,9,10,11,12,13,14,15);

    // B fragment (32x16 bf16): lane = N col; elems e -> K = half*16 + e
    v16bf bfr[4];
    bfr[0] = *(const v16bf*)(b0p + k0);
    bfr[1] = *(const v16bf*)(b1p + k0);
    bfr[2] = *(const v16bf*)(b2p + k0);
    bfr[3] = *(const v16bf*)(b3p + k0);

#pragma unroll
    for (int i = 0; i < 2; ++i)
#pragma unroll
      for (int j = 0; j < 4; ++j)
        acc[i][j] = __builtin_amdgcn_wmma_f32_16x16x32_bf16(
            false, afr[i], false, bfr[j], (short)0, acc[i][j], false, false);
  }

  // C/D layout: VGPR r -> row m0 + i*16 + half*8 + r; col = n0 + j*16 + l16
#pragma unroll
  for (int j = 0; j < 4; ++j) {
    int col = n0 + j * 16 + l16;
    if (col >= N) continue;
#pragma unroll
    for (int i = 0; i < 2; ++i) {
      int rbase = m0 + i * 16 + half * 8;
#pragma unroll
      for (int r = 0; r < 8; ++r) {
        float v = acc[i][j][r];
        size_t off = (size_t)(rbase + r) * ldc + col;
        if (EPI == 0) {
          ((__bf16*)Cout)[off] = f2bf(v);
        } else if (EPI == 1) {
          ((float*)Cout)[off] = v;
        } else {
          float t = v + bias[col];
          float s = (t > 20.0f) ? t : log1pf(__expf(t));
          ((__bf16*)Cout)[off] = f2bf(s);
        }
      }
    }
  }
}

// ---------------------------------------------------- depthwise conv3 + SiLU
__global__ __launch_bounds__(256) void conv_silu(const __bf16* __restrict__ xzb,
                                                 const float* __restrict__ w,
                                                 const float* __restrict__ bias,
                                                 __bf16* __restrict__ ub) {
  size_t idx = (size_t)blockIdx.x * 256 + threadIdx.x;   // BL*DI threads
  if (idx >= (size_t)BL * DI) return;
  int d = (int)(idx % DI);
  size_t row = idx / DI;          // b*SL + l
  int l = (int)(row & (SL - 1));
  float acc = bias[d];
  const float* wd = w + d * 3;
#pragma unroll
  for (int k = 0; k < 3; ++k) {
    int ll = l - 2 + k;
    if (ll >= 0) acc += wd[k] * bf2f(xzb[(row - 2 + k) * E2 + d]);  // xi = cols 0..DI-1
  }
  float s = acc / (1.0f + __expf(-acc));   // silu
  ub[row * DI + d] = f2bf(s);
}

// ------------------------------------------------------------ selective scan
// 1 thread per (b,d); B/C staged 64 timesteps at a time into LDS.
// Fused epilogue: y = (scan_y + u*D) * silu(z) -> bf16
__global__ __launch_bounds__(64) void scan_kernel(
    const __bf16* __restrict__ xdbl, const __bf16* __restrict__ dtb,
    const __bf16* __restrict__ ub,   const __bf16* __restrict__ xzb,
    const float* __restrict__ A_log, const float* __restrict__ Dv,
    __bf16* __restrict__ yout) {
  __shared__ float sB[64][DS];
  __shared__ float sC[64][DS];
  const int b = blockIdx.x / (DI / 64);
  const int d = (blockIdx.x % (DI / 64)) * 64 + threadIdx.x;
  const size_t rowbase = (size_t)b * SL;

  float An[DS];
#pragma unroll
  for (int n = 0; n < DS; ++n) An[n] = -__expf(A_log[d * DS + n]);
  float h[DS];
#pragma unroll
  for (int n = 0; n < DS; ++n) h[n] = 0.0f;
  const float Dd = Dv[d];

  for (int l0 = 0; l0 < SL; l0 += 64) {
    __syncthreads();
    {  // stage B (cols 48..63) and C (cols 64..79) of x_dbl, row = threadIdx.x
      const __bf16* src = xdbl + (rowbase + l0 + threadIdx.x) * XDL + DTR;
      v8bf v0 = *(const v8bf*)(src + 0);
      v8bf v1 = *(const v8bf*)(src + 8);
      v8bf v2 = *(const v8bf*)(src + 16);
      v8bf v3 = *(const v8bf*)(src + 24);
#pragma unroll
      for (int i = 0; i < 8; ++i) {
        sB[threadIdx.x][i]     = bf2f(v0[i]);
        sB[threadIdx.x][i + 8] = bf2f(v1[i]);
        sC[threadIdx.x][i]     = bf2f(v2[i]);
        sC[threadIdx.x][i + 8] = bf2f(v3[i]);
      }
    }
    __syncthreads();
    for (int t = 0; t < 64; ++t) {
      size_t row = rowbase + l0 + t;
      float dt = bf2f(dtb[row * DI + d]);
      float u  = bf2f(ub [row * DI + d]);
      float z  = bf2f(xzb[row * E2 + DI + d]);
      float du = dt * u;
      float y = 0.0f;
#pragma unroll
      for (int n = 0; n < DS; ++n) {
        float dA = __expf(dt * An[n]);
        h[n] = h[n] * dA + du * sB[t][n];
        y += h[n] * sC[t][n];
      }
      float yy = y + u * Dd;
      float sz = z / (1.0f + __expf(-z));
      yout[row * DI + d] = f2bf(yy * sz);
    }
  }
}

// ----------------------------------------------------------------- BN stats
__global__ __launch_bounds__(256) void zero_f32(float* p, int n) {
  int i = blockIdx.x * 256 + threadIdx.x;
  if (i < n) p[i] = 0.0f;
}

__global__ __launch_bounds__(256) void bn_stats(const float* __restrict__ outp,
                                                float* __restrict__ csum,
                                                float* __restrict__ csumsq,
                                                float* __restrict__ bsum) {
  int b  = blockIdx.x >> 4;
  int lc = blockIdx.x & 15;            // 16 chunks of 128 rows per batch
  size_t row0 = (size_t)b * SL + lc * 128;
  float s[3] = {0, 0, 0}, ss[3] = {0, 0, 0};
  for (int r = 0; r < 128; ++r) {
    const float* p = outp + (row0 + r) * DM;
#pragma unroll
    for (int i = 0; i < 3; ++i) {
      float v = p[threadIdx.x + i * 256];
      s[i] += v;
      ss[i] += v * v;
    }
  }
#pragma unroll
  for (int i = 0; i < 3; ++i) {
    int c = threadIdx.x + i * 256;
    atomicAdd(&csum[c], s[i]);
    atomicAdd(&csumsq[c], ss[i]);
    atomicAdd(&bsum[b * DM + c], s[i]);
  }
}

__global__ __launch_bounds__(256) void bn_finalize(const float* csum, const float* csumsq,
                                                   float* meanv, float* rsv) {
  int c = blockIdx.x * 256 + threadIdx.x;
  if (c >= DM) return;
  float m = csum[c] * (1.0f / (float)BL);
  float var = csumsq[c] * (1.0f / (float)BL) - m * m;
  meanv[c] = m;
  rsv[c] = rsqrtf(var + BN_EPS);
}

__global__ __launch_bounds__(256) void pooled_kernel(const float* bsum, const float* meanv,
                                                     const float* rsv, const float* gamma,
                                                     const float* beta, float* pooled) {
  int idx = blockIdx.x * 256 + threadIdx.x;   // NB*DM
  if (idx >= NB * DM) return;
  int c = idx % DM;
  pooled[idx] = (bsum[idx] * (1.0f / (float)SL) - meanv[c]) * rsv[c] * gamma[c] + beta[c];
}

// ----------------------------------------------------------------------- SE
__global__ __launch_bounds__(64) void se_kernel(const float* __restrict__ pooled,
                                                const float* __restrict__ w1,
                                                const float* __restrict__ w2,
                                                float* __restrict__ g) {
  __shared__ float sp[DM];
  __shared__ float sh[DM / 16];
  int b = blockIdx.x;
  for (int c = threadIdx.x; c < DM; c += 64) sp[c] = pooled[b * DM + c];
  __syncthreads();
  if (threadIdx.x < DM / 16) {
    const float* w = w1 + threadIdx.x * DM;
    float a = 0.0f;
    for (int c = 0; c < DM; ++c) a += sp[c] * w[c];
    sh[threadIdx.x] = fmaxf(a, 0.0f);
  }
  __syncthreads();
  for (int c = threadIdx.x; c < DM; c += 64) {
    const float* w = w2 + c * (DM / 16);
    float a = 0.0f;
#pragma unroll
    for (int j = 0; j < DM / 16; ++j) a += sh[j] * w[j];
    g[b * DM + c] = 1.0f / (1.0f + __expf(-a));
  }
}

// --------------------------------------------------------- final BN * gate
__global__ __launch_bounds__(256) void final_kernel(const float* __restrict__ outp,
                                                    const float* __restrict__ meanv,
                                                    const float* __restrict__ rsv,
                                                    const float* __restrict__ gamma,
                                                    const float* __restrict__ beta,
                                                    const float* __restrict__ g,
                                                    float* __restrict__ dout) {
  size_t idx = (size_t)blockIdx.x * 256 + threadIdx.x;   // BL*DM
  if (idx >= (size_t)BL * DM) return;
  int c = (int)(idx % DM);
  size_t row = idx / DM;
  int b = (int)(row >> 11);
  float v = (outp[idx] - meanv[c]) * rsv[c] * gamma[c] + beta[c];
  dout[idx] = v * g[b * DM + c];
}

// ================================================================== launch
extern "C" void kernel_launch(void* const* d_in, const int* in_sizes, int n_in,
                              void* d_out, int out_size, void* d_ws, size_t ws_size,
                              hipStream_t stream) {
  const float* x         = (const float*)d_in[0];
  const float* in_proj_w = (const float*)d_in[1];
  const float* conv_w    = (const float*)d_in[2];
  const float* conv_b    = (const float*)d_in[3];
  const float* x_proj_w  = (const float*)d_in[4];
  const float* dt_proj_w = (const float*)d_in[5];
  const float* dt_proj_b = (const float*)d_in[6];
  const float* A_log     = (const float*)d_in[7];
  const float* Dv        = (const float*)d_in[8];
  const float* out_projw = (const float*)d_in[9];
  const float* bn_gamma  = (const float*)d_in[10];
  const float* bn_beta   = (const float*)d_in[11];
  const float* se_w1     = (const float*)d_in[12];
  const float* se_w2     = (const float*)d_in[13];
  (void)in_sizes; (void)n_in; (void)out_size; (void)ws_size;

  char* ws = (char*)d_ws;
  size_t off = 0;
  auto alloc = [&](size_t bytes) -> char* {
    char* p = ws + off;
    off += (bytes + 255) & ~(size_t)255;
    return p;
  };

  __bf16* x_bf   = (__bf16*)alloc((size_t)BL * DM * 2);
  __bf16* w_in   = (__bf16*)alloc((size_t)E2 * DM * 2);
  __bf16* w_xp   = (__bf16*)alloc((size_t)NXPP * DI * 2);   // padded to 128 rows
  __bf16* w_dt   = (__bf16*)alloc((size_t)DI * KPAD * 2);
  __bf16* w_out  = (__bf16*)alloc((size_t)DM * DI * 2);
  __bf16* xz_bf  = (__bf16*)alloc((size_t)BL * E2 * 2);
  __bf16* u_bf   = (__bf16*)alloc((size_t)BL * DI * 2);
  __bf16* xdbl   = (__bf16*)alloc((size_t)BL * XDL * 2);
  __bf16* dt_bf  = (__bf16*)alloc((size_t)BL * DI * 2);
  __bf16* y_bf   = (__bf16*)alloc((size_t)BL * DI * 2);
  float*  out_f  = (float*)alloc((size_t)BL * DM * 4);
  float*  csum   = (float*)alloc(DM * 4);            // contiguous with next two
  float*  csumsq = (float*)alloc(DM * 4);
  float*  bsum   = (float*)alloc(NB * DM * 4);
  float*  meanv  = (float*)alloc(DM * 4);
  float*  rsv    = (float*)alloc(DM * 4);
  float*  pooled = (float*)alloc(NB * DM * 4);
  float*  gbuf   = (float*)alloc(NB * DM * 4);

  // ---- fp32 -> bf16 staging
  cvt_f32_bf16<<<(int)(((size_t)BL * DM + 255) / 256), 256, 0, stream>>>(x, x_bf, (size_t)BL * DM);
  cvt_f32_bf16<<<(E2 * DM + 255) / 256, 256, 0, stream>>>(in_proj_w, w_in, (size_t)E2 * DM);
  cvt_pad_xpw<<<(NXPP * DI) / 256, 256, 0, stream>>>(x_proj_w, w_xp);
  cvt_pad_dtw<<<(DI * KPAD) / 256, 256, 0, stream>>>(dt_proj_w, w_dt);
  cvt_f32_bf16<<<(DM * DI + 255) / 256, 256, 0, stream>>>(out_projw, w_out, (size_t)DM * DI);

  // ---- xz = x @ in_proj_w^T        (M=32768, N=3072, K=768)
  gemm_bf16_wmma<0><<<dim3(BL / 128, E2 / 128), 256, 0, stream>>>(
      x_bf, w_in, xz_bf, nullptr, BL, E2, DM, DM, DM, E2);

  // ---- depthwise causal conv3 + bias + silu -> u
  conv_silu<<<(int)(((size_t)BL * DI + 255) / 256), 256, 0, stream>>>(xz_bf, conv_w, conv_b, u_bf);

  // ---- x_dbl = u @ x_proj_w^T      (M=32768, N=80 of 128 padded, K=1536)
  gemm_bf16_wmma<0><<<dim3(BL / 128, 1), 256, 0, stream>>>(
      u_bf, w_xp, xdbl, nullptr, BL, NXP, DI, DI, DI, XDL);

  // ---- dt = softplus(dt_r @ dt_proj_w^T + b)  (K padded 48->64, zero weights)
  gemm_bf16_wmma<2><<<dim3(BL / 128, DI / 128), 256, 0, stream>>>(
      xdbl, w_dt, dt_bf, dt_proj_b, BL, DI, KPAD, XDL, KPAD, DI);

  // ---- selective scan + fused (y + u*D) * silu(z)
  scan_kernel<<<NB * (DI / 64), 64, 0, stream>>>(xdbl, dt_bf, u_bf, xz_bf, A_log, Dv, y_bf);

  // ---- out = y @ out_proj_w^T      (M=32768, N=768, K=1536), fp32 out
  gemm_bf16_wmma<1><<<dim3(BL / 128, DM / 128), 256, 0, stream>>>(
      y_bf, w_out, out_f, nullptr, BL, DM, DI, DI, DI, DM);

  // ---- batch-norm stats (+ per-batch sums for pooling)
  zero_f32<<<((2 + NB) * DM + 255) / 256, 256, 0, stream>>>(csum, (2 + NB) * DM);
  bn_stats<<<NB * 16, 256, 0, stream>>>(out_f, csum, csumsq, bsum);
  bn_finalize<<<(DM + 255) / 256, 256, 0, stream>>>(csum, csumsq, meanv, rsv);
  pooled_kernel<<<(NB * DM + 255) / 256, 256, 0, stream>>>(bsum, meanv, rsv, bn_gamma, bn_beta, pooled);

  // ---- squeeze-excite gate
  se_kernel<<<NB, 64, 0, stream>>>(pooled, se_w1, se_w2, gbuf);

  // ---- final: xn * g
  final_kernel<<<(int)(((size_t)BL * DM + 255) / 256), 256, 0, stream>>>(
      out_f, meanv, rsv, bn_gamma, bn_beta, gbuf, (float*)d_out);
}
